// SelfContrastiveLoss_49297634624123
// MI455X (gfx1250) — compile-verified, matrix operand order checked by reading, not run on hardware
//
#include <hip/hip_runtime.h>

// SelfContrastiveLoss for MI455X (gfx1250, wave32, WMMA, async LDS copies).
// B=8192, D=256, TEMP=0.05, EPS=1e-5, NORM_EPS=1e-12.
// den_qk = full row sum of E, den_kq = full col sum of E, d = diag(E).

#define BDIM   8192
#define DDIM   256
#define INV_T  20.0f
#define LEPS   1e-5f
#define NEPS   1e-12f
#define LDS_STRIDE 40   // 32 halfs + 8 pad -> 80B row stride, conflict-free b128 lane reads

typedef __attribute__((ext_vector_type(16))) _Float16 v16h;
typedef __attribute__((ext_vector_type(8)))  float    v8f;
typedef __attribute__((ext_vector_type(4)))  int      v4i;

typedef __attribute__((address_space(1))) v4i gv4i;   // global
typedef __attribute__((address_space(3))) v4i lv4i;   // LDS

union Frag16 {
    v16h  v;
    uint4 u[2];
};

struct H4 { _Float16 h[4]; };

// ---- gfx1250 async global->LDS copy (ASYNCcnt path), with safe fallback ----
#if defined(__HIP_DEVICE_COMPILE__) && __has_builtin(__builtin_amdgcn_global_load_async_to_lds_b128)
#define HAS_ASYNC_LDS 1
#else
#define HAS_ASYNC_LDS 0
#endif

__device__ __forceinline__ void async_copy_b128(const _Float16* gsrc, _Float16* ldst) {
#if HAS_ASYNC_LDS
    __builtin_amdgcn_global_load_async_to_lds_b128(
        (gv4i*)gsrc, (lv4i*)ldst, /*offset=*/0, /*cpol=*/0);
#else
    *(uint4*)ldst = *(const uint4*)gsrc;
#endif
}

__device__ __forceinline__ void wait_async_done() {
#if HAS_ASYNC_LDS
#if __has_builtin(__builtin_amdgcn_s_wait_asynccnt)
    __builtin_amdgcn_s_wait_asynccnt(0);
#else
    asm volatile("s_wait_asynccnt 0x0" ::: "memory");
#endif
#endif
}

__device__ __forceinline__ void store_h4(_Float16* dst, float4 v, float s) {
    H4 o;
    o.h[0] = (_Float16)(v.x * s);
    o.h[1] = (_Float16)(v.y * s);
    o.h[2] = (_Float16)(v.z * s);
    o.h[3] = (_Float16)(v.w * s);
    *(H4*)dst = o;
}

// ---------------------------------------------------------------------------
// Kernel 1: L2-normalize rows of q and k -> f16, zero the sum accumulators.
// One wave (32 lanes) per row; 8 rows per 256-thread block.
// ---------------------------------------------------------------------------
__global__ __launch_bounds__(256) void scl_norm_kernel(
    const float* __restrict__ q, const float* __restrict__ k,
    _Float16* __restrict__ qh, _Float16* __restrict__ kh,
    float* __restrict__ rowSum, float* __restrict__ colSum)
{
    const int wave = threadIdx.x >> 5;
    const int lane = threadIdx.x & 31;
    const int row  = blockIdx.x * 8 + wave;

    const float4* qr = (const float4*)(q + (size_t)row * DDIM);
    const float4* kr = (const float4*)(k + (size_t)row * DDIM);
    float4 q0 = qr[lane], q1 = qr[lane + 32];
    float4 k0 = kr[lane], k1 = kr[lane + 32];

    float sq = q0.x*q0.x + q0.y*q0.y + q0.z*q0.z + q0.w*q0.w
             + q1.x*q1.x + q1.y*q1.y + q1.z*q1.z + q1.w*q1.w;
    float sk = k0.x*k0.x + k0.y*k0.y + k0.z*k0.z + k0.w*k0.w
             + k1.x*k1.x + k1.y*k1.y + k1.z*k1.z + k1.w*k1.w;

#pragma unroll
    for (int off = 1; off < 32; off <<= 1) {
        sq += __shfl_xor(sq, off, 32);
        sk += __shfl_xor(sk, off, 32);
    }

    const float rq = 1.0f / fmaxf(sqrtf(sq), NEPS);
    const float rk = 1.0f / fmaxf(sqrtf(sk), NEPS);

    _Float16* qd = qh + (size_t)row * DDIM;
    _Float16* kd = kh + (size_t)row * DDIM;
    store_h4(qd + lane * 4,        q0, rq);
    store_h4(qd + 128 + lane * 4,  q1, rq);
    store_h4(kd + lane * 4,        k0, rk);
    store_h4(kd + 128 + lane * 4,  k1, rk);

    if (lane == 0) {
        rowSum[row] = 0.0f;
        colSum[row] = 0.0f;
    }
}

// ---------------------------------------------------------------------------
// Kernel 2: tiled 128x128 WMMA f16 GEMM (S = qn . kn^T), double-buffered
// async global->LDS staging, fused exp epilogue, diagonal capture, atomic
// row/col sum accumulation. 8 waves/block (4M x 2N); each wave owns a 2x4
// grid of 16x16 f32 accumulators (32 rows x 64 cols).
// ---------------------------------------------------------------------------
__global__ __launch_bounds__(256) void scl_gemm_exp_kernel(
    const _Float16* __restrict__ qh, const _Float16* __restrict__ kh,
    float* __restrict__ diag, float* __restrict__ rowSum, float* __restrict__ colSum)
{
    __shared__ _Float16 ldsA[2][128 * LDS_STRIDE];
    __shared__ _Float16 ldsB[2][128 * LDS_STRIDE];

    const int tid   = threadIdx.x;
    const int lane  = tid & 31;
    const int wave  = tid >> 5;
    const int waveM = wave >> 1;   // 0..3 -> 32-row strip
    const int waveN = wave & 1;    // 0..1 -> 64-col strip
    const int lmod  = lane & 15;
    const int half16 = lane >> 4;  // 0: lanes 0-15, 1: lanes 16-31
    const int koff  = half16 ? 8 : 0;   // ISA 16-bit A/B fragment K interleave

    const int tileI = blockIdx.x;  // q-row tile
    const int tileJ = blockIdx.y;  // k-row tile

    // cooperative staging mapping: 2 threads per row, 16 halfs each
    const int loadRow  = tid >> 1;
    const int loadHalf = (tid & 1) * 16;
    const _Float16* gq = qh + ((size_t)tileI * 128 + loadRow) * DDIM + loadHalf;
    const _Float16* gk = kh + ((size_t)tileJ * 128 + loadRow) * DDIM + loadHalf;
    const int ldOff = loadRow * LDS_STRIDE + loadHalf;

    v8f acc[2][4] = {};

    // prologue: stage K-chunk 0 into buffer 0
    async_copy_b128(gq,     &ldsA[0][ldOff]);
    async_copy_b128(gq + 8, &ldsA[0][ldOff + 8]);
    async_copy_b128(gk,     &ldsB[0][ldOff]);
    async_copy_b128(gk + 8, &ldsB[0][ldOff + 8]);

    for (int it = 0; it < DDIM / 32; ++it) {
        const int cur = it & 1;
        wait_async_done();      // this wave's copies into buf[cur] have landed
        __syncthreads();        // everyone's copies landed; buf[cur^1] reads from it-1 done

        if (it + 1 < DDIM / 32) {
            const int kk = (it + 1) * 32;
            const int nxt = cur ^ 1;
            async_copy_b128(gq + kk,     &ldsA[nxt][ldOff]);
            async_copy_b128(gq + kk + 8, &ldsA[nxt][ldOff + 8]);
            async_copy_b128(gk + kk,     &ldsB[nxt][ldOff]);
            async_copy_b128(gk + kk + 8, &ldsB[nxt][ldOff + 8]);
        }

        const _Float16* bufA = ldsA[cur];
        const _Float16* bufB = ldsB[cur];

        Frag16 af[2], bf[4];
#pragma unroll
        for (int m = 0; m < 2; ++m) {
            const _Float16* p = bufA + (waveM * 32 + m * 16 + lmod) * LDS_STRIDE + koff;
            af[m].u[0] = *(const uint4*)(p);
            af[m].u[1] = *(const uint4*)(p + 16);
        }
#pragma unroll
        for (int n = 0; n < 4; ++n) {
            const _Float16* p = bufB + (waveN * 64 + n * 16 + lmod) * LDS_STRIDE + koff;
            bf[n].u[0] = *(const uint4*)(p);
            bf[n].u[1] = *(const uint4*)(p + 16);
        }
#pragma unroll
        for (int m = 0; m < 2; ++m)
#pragma unroll
            for (int n = 0; n < 4; ++n)
                acc[m][n] = __builtin_amdgcn_wmma_f32_16x16x32_f16(
                    /*neg_a=*/false, af[m].v, /*neg_b=*/false, bf[n].v,
                    /*c_mod=*/(short)0, acc[m][n],
                    /*reuse_a=*/false, /*reuse_b=*/false);
    }

    // Epilogue: E = exp(S/T); diagonal capture; row/col partial sums.
    // C/D layout: vgpr v, lanes 0-15 -> row v cols 0-15; lanes 16-31 -> row v+8.
    const int rowBase = tileI * 128 + waveM * 32;
    const int colBase = tileJ * 128 + waveN * 64;

    float rsum[2][8];
#pragma unroll
    for (int m = 0; m < 2; ++m)
#pragma unroll
        for (int v = 0; v < 8; ++v) rsum[m][v] = 0.0f;
    float csum[4] = {0.0f, 0.0f, 0.0f, 0.0f};

#pragma unroll
    for (int m = 0; m < 2; ++m) {
#pragma unroll
        for (int n = 0; n < 4; ++n) {
            const v8f a = acc[m][n];
#pragma unroll
            for (int v = 0; v < 8; ++v) {
                const float e = __expf(a[v] * INV_T);
                rsum[m][v] += e;
                csum[n]    += e;
                const int gi = rowBase + m * 16 + v + half16 * 8;
                const int gj = colBase + n * 16 + lmod;
                if (gi == gj) diag[gi] = e;
            }
        }
    }

    // Row sums: reduce across the 16 lanes of each half-wave.
#pragma unroll
    for (int m = 0; m < 2; ++m) {
#pragma unroll
        for (int v = 0; v < 8; ++v) {
            float r = rsum[m][v];
            r += __shfl_xor(r, 1, 32);
            r += __shfl_xor(r, 2, 32);
            r += __shfl_xor(r, 4, 32);
            r += __shfl_xor(r, 8, 32);
            if (lmod == 0)
                atomicAdd(&rowSum[rowBase + m * 16 + v + half16 * 8], r);
        }
    }

    // Col sums: lanes L and L+16 hold the same column.
#pragma unroll
    for (int n = 0; n < 4; ++n) {
        float c = csum[n];
        c += __shfl_xor(c, 16, 32);
        if (lane < 16)
            atomicAdd(&colSum[colBase + n * 16 + lane], c);
    }
}

// ---------------------------------------------------------------------------
// Kernel 3: final scalar loss. Single 256-thread block.
// loss = (sum -log(d/rowSum + eps) + sum -log(d/colSum + eps)) / B
// ---------------------------------------------------------------------------
__global__ __launch_bounds__(256) void scl_loss_kernel(
    const float* __restrict__ diag, const float* __restrict__ rowSum,
    const float* __restrict__ colSum, float* __restrict__ out)
{
    __shared__ float red[8];
    float acc = 0.0f;
    for (int i = threadIdx.x; i < BDIM; i += 256) {
        const float d = diag[i];
        acc -= __logf(d / rowSum[i] + LEPS);
        acc -= __logf(d / colSum[i] + LEPS);
    }
#pragma unroll
    for (int off = 1; off < 32; off <<= 1)
        acc += __shfl_xor(acc, off, 32);

    const int lane = threadIdx.x & 31;
    const int wave = threadIdx.x >> 5;
    if (lane == 0) red[wave] = acc;
    __syncthreads();
    if (threadIdx.x == 0) {
        float t = 0.0f;
#pragma unroll
        for (int w = 0; w < 8; ++w) t += red[w];
        out[0] = t / (float)BDIM;
    }
}

// ---------------------------------------------------------------------------
extern "C" void kernel_launch(void* const* d_in, const int* in_sizes, int n_in,
                              void* d_out, int out_size, void* d_ws, size_t ws_size,
                              hipStream_t stream)
{
    (void)in_sizes; (void)n_in; (void)out_size; (void)ws_size;

    const float* q = (const float*)d_in[0];
    const float* k = (const float*)d_in[1];
    float* out = (float*)d_out;

    // workspace layout
    _Float16* qh     = (_Float16*)d_ws;                       // 4 MiB
    _Float16* kh     = qh + (size_t)BDIM * DDIM;              // 4 MiB
    float*    diag   = (float*)(kh + (size_t)BDIM * DDIM);    // 32 KiB
    float*    rowSum = diag + BDIM;                           // 32 KiB
    float*    colSum = rowSum + BDIM;                         // 32 KiB

    scl_norm_kernel<<<BDIM / 8, 256, 0, stream>>>(q, k, qh, kh, rowSum, colSum);
    scl_gemm_exp_kernel<<<dim3(BDIM / 128, BDIM / 128), 256, 0, stream>>>(
        qh, kh, diag, rowSum, colSum);
    scl_loss_kernel<<<1, 256, 0, stream>>>(diag, rowSum, colSum, out);
}
